// Attention_23098334118519
// MI455X (gfx1250) — matrix operand order repaired and checked by reading, compiled, not run on hardware
//
#include <hip/hip_runtime.h>
#include <hip/hip_bf16.h>
#include <math.h>

// Problem constants (B=2, S=2048, D=2048, H=32, KVH=8, DH=64, GROUPS=4)
#define BATCH 2
#define SEQ   2048
#define DMODEL 2048
#define NH    32
#define NKVH  8
#define DHEAD 64
#define GROUPS 4

typedef __attribute__((ext_vector_type(16))) __bf16 v16bf;
typedef __attribute__((ext_vector_type(8)))  float  v8f;

#define WMMA_BF16(A, B, C) \
  __builtin_amdgcn_wmma_f32_16x16x32_bf16(false, (A), false, (B), (short)0, (C), false, false)

// CDNA5 async global->LDS copy (16B per lane, tracked by ASYNCcnt).
// LDS byte offset = low 32 bits of the generic pointer (aperture|offset form).
__device__ __forceinline__ void async_copy_b128(const float* g, float* lds) {
  unsigned lo = (unsigned)(size_t)lds;
  asm volatile("global_load_async_to_lds_b128 %0, %1, off"
               :: "v"(lo), "v"(g) : "memory");
}
__device__ __forceinline__ void wait_asynccnt0() {
  asm volatile("s_wait_asynccnt 0x0" ::: "memory");
}
__device__ __forceinline__ void wait_dscnt0() {
  asm volatile("s_wait_dscnt 0x0" ::: "memory");
}

// Reductions within each 16-lane half of a wave32 (xor of bits 0..3 never
// crosses the half boundary).
__device__ __forceinline__ float red_max16(float v) {
  v = fmaxf(v, __shfl_xor(v, 1, 32));
  v = fmaxf(v, __shfl_xor(v, 2, 32));
  v = fmaxf(v, __shfl_xor(v, 4, 32));
  v = fmaxf(v, __shfl_xor(v, 8, 32));
  return v;
}
__device__ __forceinline__ float red_sum16(float v) {
  v += __shfl_xor(v, 1, 32);
  v += __shfl_xor(v, 2, 32);
  v += __shfl_xor(v, 4, 32);
  v += __shfl_xor(v, 8, 32);
  return v;
}

// ---------------------------------------------------------------------------
// Generic GEMM:  C[M,N] = A[M,K] * W[N,K]^T   (fp32 memory, bf16 WMMA)
// workgroup tile 128x128, BK=32, 8 waves (4x2), 32x64 per wave (2x4 WMMA tiles)
// Arithmetic intensity: 2*128*128*32 FLOP per 32KB global bytes = 32 F/B.
// ---------------------------------------------------------------------------
__global__ void __launch_bounds__(256)
gemm_xWt(const float* __restrict__ A, const float* __restrict__ W,
         float* __restrict__ C, int M, int N, int K)
{
  __shared__ __bf16 As[128][40];  // 32 cols + 8 pad (kills A-frag bank conflicts)
  __shared__ __bf16 Ws[128][40];

  const int tid  = threadIdx.x;
  const int lane = tid & 31;
  const int wave = tid >> 5;
  const int wm   = wave >> 1;      // 0..3 (M direction, 32 rows each)
  const int wn   = wave & 1;       // 0..1 (N direction, 64 cols each)
  const int lm   = lane & 15;
  const int hi   = lane >> 4;
  const int m0   = blockIdx.y * 128;
  const int n0   = blockIdx.x * 128;

  const int srow = tid >> 3;        // staging row 0..31 (first slot)
  const int sc4  = (tid & 7) << 2;  // staging col (float4)

  const v8f vz = {};
  v8f acc[2][4];
  #pragma unroll
  for (int i = 0; i < 2; ++i)
    #pragma unroll
    for (int j = 0; j < 4; ++j) acc[i][j] = vz;

  for (int kk = 0; kk < K; kk += 32) {
    // prefetch next K-slab (global_prefetch_b8; one line hint per thread)
    if (kk + 32 < K) {
      __builtin_prefetch(A + (size_t)(m0 + srow) * K + kk + 32 + sc4, 0, 1);
      __builtin_prefetch(W + (size_t)(n0 + srow) * K + kk + 32 + sc4, 0, 1);
    }
    // ---- stage A tile 128x32 (1024 float4, 4/thread), convert to bf16
    #pragma unroll
    for (int it = 0; it < 4; ++it) {
      int idx = tid + it * 256;
      int row = idx >> 3;
      int c4  = (idx & 7) << 2;
      const float4 v = *reinterpret_cast<const float4*>(
          A + (size_t)(m0 + row) * K + kk + c4);
      As[row][c4 + 0] = (__bf16)v.x;  As[row][c4 + 1] = (__bf16)v.y;
      As[row][c4 + 2] = (__bf16)v.z;  As[row][c4 + 3] = (__bf16)v.w;
    }
    // ---- stage W tile 128x32 (1024 float4, 4/thread)
    #pragma unroll
    for (int it = 0; it < 4; ++it) {
      int idx = tid + it * 256;
      int row = idx >> 3;
      int c4  = (idx & 7) << 2;
      const float4 v = *reinterpret_cast<const float4*>(
          W + (size_t)(n0 + row) * K + kk + c4);
      Ws[row][c4 + 0] = (__bf16)v.x;  Ws[row][c4 + 1] = (__bf16)v.y;
      Ws[row][c4 + 2] = (__bf16)v.z;  Ws[row][c4 + 3] = (__bf16)v.w;
    }
    __syncthreads();

    // ---- A fragments (16x32, ISA layout: lanes0-15 K{0..7,16..23}, lanes16-31 K{8..15,24..31})
    v16bf a[2], b[4];
    #pragma unroll
    for (int i = 0; i < 2; ++i) {
      int r = wm * 32 + i * 16 + lm;
      #pragma unroll
      for (int t = 0; t < 8; ++t) {
        a[i][t]     = As[r][hi * 8 + t];
        a[i][t + 8] = As[r][16 + hi * 8 + t];
      }
    }
    // ---- B fragments (32x16, lanes0-15 K{0..7,16..23}, lanes16-31 K{8..15,24..31})
    #pragma unroll
    for (int j = 0; j < 4; ++j) {
      int c = wn * 64 + j * 16 + lm;
      #pragma unroll
      for (int t = 0; t < 16; ++t) b[j][t] = Ws[c][(t < 8 ? hi * 8 + t : 16 + hi * 8 + (t - 8))];
    }

    #pragma unroll
    for (int i = 0; i < 2; ++i)
      #pragma unroll
      for (int j = 0; j < 4; ++j)
        acc[i][j] = WMMA_BF16(a[i], b[j], acc[i][j]);
    __syncthreads();
  }

  // ---- store C (C-frag layout: row = r + 8*hi, col = lane&15)
  #pragma unroll
  for (int i = 0; i < 2; ++i)
    #pragma unroll
    for (int j = 0; j < 4; ++j) {
      int col = n0 + wn * 64 + j * 16 + lm;
      #pragma unroll
      for (int r = 0; r < 8; ++r) {
        int row = m0 + wm * 32 + i * 16 + r + 8 * hi;
        C[(size_t)row * N + col] = acc[i][j][r];
      }
    }
}

// ---------------------------------------------------------------------------
// RoPE (in place): buf is (rows, heads*64); pair i uses freqs[(s*32+i)*2 +{0,1}]
// ---------------------------------------------------------------------------
__global__ void __launch_bounds__(256)
rope_kernel(float* __restrict__ buf, const float* __restrict__ freqs,
            int rows, int heads)
{
  int idx = blockIdx.x * blockDim.x + threadIdx.x;
  int total = rows * heads * (DHEAD / 2);
  if (idx >= total) return;
  int i   = idx & 31;          // pair index 0..31
  int t   = idx >> 5;
  int h   = t % heads;
  int row = t / heads;
  int s   = row & (SEQ - 1);
  float c  = freqs[(s * 32 + i) * 2 + 0];
  float sn = freqs[(s * 32 + i) * 2 + 1];
  float* p = buf + (size_t)row * heads * DHEAD + h * DHEAD + 2 * i;
  float x1 = p[0], x2 = p[1];
  p[0] = x1 * c - x2 * sn;
  p[1] = x1 * sn + x2 * c;
}

// ---------------------------------------------------------------------------
// Fused flash attention (causal, GQA). 2 waves/block; each wave owns one
// 16-query tile with its own causal trip count, so synchronization is
// wave-local (s_wait_asynccnt / s_wait_dscnt), never a block barrier.
// K/V chunks staged fp32 via async global->LDS DMA; bf16 conversion happens
// in the per-fragment gather.
// Q: (B*S, 2048) col h*64+d ; Kp/Vp: (B*S, 512) col kvh*64+d
// ---------------------------------------------------------------------------
__global__ void __launch_bounds__(64)
attn_fused(const float* __restrict__ Q, const float* __restrict__ Kp,
           const float* __restrict__ Vp, float* __restrict__ AO)
{
  __shared__ float  Ksf[2][32][68];   // 32 keys x 64 dims fp32 (+4 pad)
  __shared__ float  Vsf[2][32][68];
  __shared__ __bf16 Ps [2][16][40];   // 16 q x 32 keys bf16 (+8 pad)

  const int lane  = threadIdx.x & 31;
  const int wave  = threadIdx.x >> 5;
  const int b     = blockIdx.x >> 5;          // H = 32
  const int h     = blockIdx.x & 31;
  const int kvh   = h >> 2;                   // GROUPS = 4
  const int qbase = blockIdx.y * 32 + wave * 16;
  const int lm    = lane & 15;
  const int hi    = lane >> 4;

  // ---- Q fragment (16x64 -> two 16x32 A-frags), loaded straight from global
  const float* qrow = Q + ((size_t)(b * SEQ + qbase + lm)) * DMODEL + h * DHEAD;
  v16bf qa[2];
  #pragma unroll
  for (int kh = 0; kh < 2; ++kh) {
    int db = kh * 32;
    #pragma unroll
    for (int t = 0; t < 8; ++t) {
      qa[kh][t]     = (__bf16)qrow[db + hi * 8 + t];
      qa[kh][t + 8] = (__bf16)qrow[db + 16 + hi * 8 + t];
    }
  }

  const v8f vz = {};
  v8f oacc[4];
  #pragma unroll
  for (int d = 0; d < 4; ++d) oacc[d] = vz;
  float mrun[8], lrun[8];
  #pragma unroll
  for (int r = 0; r < 8; ++r) { mrun[r] = -1e30f; lrun[r] = 0.f; }

  const int nch = (qbase + 16 + 31) >> 5;     // causal: only chunks <= diagonal
  for (int ch = 0; ch < nch; ++ch) {
    const int t0 = ch * 32;

    // ---- async-stage K/V chunk (32x64 fp32 each; 16B per lane per issue)
    const float* kb = Kp + ((size_t)(b * SEQ + t0)) * (NKVH * DHEAD) + kvh * DHEAD;
    const float* vb = Vp + ((size_t)(b * SEQ + t0)) * (NKVH * DHEAD) + kvh * DHEAD;
    #pragma unroll
    for (int it = 0; it < 16; ++it) {
      int idx = lane + it * 32;
      int row = idx >> 4;
      int c4  = (idx & 15) << 2;
      async_copy_b128(kb + (size_t)row * (NKVH * DHEAD) + c4, &Ksf[wave][row][c4]);
      async_copy_b128(vb + (size_t)row * (NKVH * DHEAD) + c4, &Vsf[wave][row][c4]);
    }
    wait_asynccnt0();

    // ---- scores: 16 q x 32 keys = 2 N-tiles, K-dim = 64 -> 2 WMMAs each
    v8f sc[2]; sc[0] = vz; sc[1] = vz;
    #pragma unroll
    for (int f = 0; f < 2; ++f) {
      #pragma unroll
      for (int kh = 0; kh < 2; ++kh) {
        v16bf bf;   // B[k=d][n=key] = K[key][d], fp32->bf16 at gather
        #pragma unroll
        for (int t = 0; t < 16; ++t)
          bf[t] = (__bf16)Ksf[wave][f * 16 + lm][kh * 32 + (t < 8 ? hi * 8 + t : 16 + hi * 8 + (t - 8))];
        sc[f] = WMMA_BF16(qa[kh], bf, sc[f]);
      }
    }

    // ---- online softmax (row = r + 8*hi, key = lm + 16*f + t0)
    float rm[8], alpha[8];
    #pragma unroll
    for (int r = 0; r < 8; ++r) {
      int qg = qbase + r + 8 * hi;
      float s0 = sc[0][r]; if (t0 + lm > qg)      s0 = -1e9f;
      float s1 = sc[1][r]; if (t0 + 16 + lm > qg) s1 = -1e9f;
      sc[0][r] = s0; sc[1][r] = s1;
      rm[r] = red_max16(fmaxf(s0, s1));
    }
    #pragma unroll
    for (int r = 0; r < 8; ++r) {
      float mnew = fmaxf(mrun[r], rm[r]);
      alpha[r] = __expf(mrun[r] - mnew);
      mrun[r]  = mnew;
      float p0 = __expf(sc[0][r] - mnew);
      float p1 = __expf(sc[1][r] - mnew);
      Ps[wave][r + 8 * hi][lm]      = (__bf16)p0;   // C-layout -> (m,k) in LDS
      Ps[wave][r + 8 * hi][16 + lm] = (__bf16)p1;
      lrun[r] = lrun[r] * alpha[r] + red_sum16(p0 + p1);
    }
    #pragma unroll
    for (int d = 0; d < 4; ++d)
      #pragma unroll
      for (int r = 0; r < 8; ++r) oacc[d][r] *= alpha[r];

    wait_dscnt0();                      // Ps visible to this wave

    // ---- P(16x32) x V(32x64): reload P in A-layout, 4 N-tiles of V
    v16bf pa;
    #pragma unroll
    for (int t = 0; t < 8; ++t) {
      pa[t]     = Ps[wave][lm][hi * 8 + t];
      pa[t + 8] = Ps[wave][lm][16 + hi * 8 + t];
    }
    #pragma unroll
    for (int d = 0; d < 4; ++d) {
      v16bf vbf;   // B[k=key][n=d] = V[key][d], fp32->bf16 at gather
      #pragma unroll
      for (int t = 0; t < 16; ++t)
        vbf[t] = (__bf16)Vsf[wave][(t < 8 ? hi * 8 + t : 16 + hi * 8 + (t - 8))][d * 16 + lm];
      oacc[d] = WMMA_BF16(pa, vbf, oacc[d]);
    }
    wait_dscnt0();                      // LDS reads retired before restaging
  }

  // ---- normalize and write attn output (row-major (B*S, H*64))
  #pragma unroll
  for (int d = 0; d < 4; ++d)
    #pragma unroll
    for (int r = 0; r < 8; ++r) {
      int row = qbase + r + 8 * hi;
      AO[((size_t)(b * SEQ + row)) * DMODEL + h * DHEAD + d * 16 + lm] =
          oacc[d][r] / lrun[r];
    }
}

// ---------------------------------------------------------------------------
extern "C" void kernel_launch(void* const* d_in, const int* in_sizes, int n_in,
                              void* d_out, int out_size, void* d_ws, size_t ws_size,
                              hipStream_t stream)
{
  const float* x    = (const float*)d_in[0];  // (B,S,D)
  const float* rope = (const float*)d_in[1];  // (S, 32, 2)
  // d_in[2] = additive causal mask (unused; applied analytically)
  const float* Wq = (const float*)d_in[3];    // (2048, 2048)
  const float* Wk = (const float*)d_in[4];    // (512, 2048)
  const float* Wv = (const float*)d_in[5];    // (512, 2048)
  const float* Wo = (const float*)d_in[6];    // (2048, 2048)
  float* out = (float*)d_out;

  const size_t BS = (size_t)BATCH * SEQ;      // 4096
  float* Qb = (float*)d_ws;                   // (4096, 2048)
  float* Kb = Qb + BS * DMODEL;               // (4096, 512)
  float* Vb = Kb + BS * (NKVH * DHEAD);       // (4096, 512)
  float* AO = Vb + BS * (NKVH * DHEAD);       // (4096, 2048)  (~84 MB total)

  // QKV projections (bf16 WMMA, fp32 accumulate)
  gemm_xWt<<<dim3(DMODEL / 128, BS / 128), 256, 0, stream>>>(x, Wq, Qb, BS, DMODEL, DMODEL);
  gemm_xWt<<<dim3((NKVH * DHEAD) / 128, BS / 128), 256, 0, stream>>>(x, Wk, Kb, BS, NKVH * DHEAD, DMODEL);
  gemm_xWt<<<dim3((NKVH * DHEAD) / 128, BS / 128), 256, 0, stream>>>(x, Wv, Vb, BS, NKVH * DHEAD, DMODEL);

  // RoPE on Q and K
  rope_kernel<<<(int)((BS * NH * 32 + 255) / 256), 256, 0, stream>>>(Qb, rope, (int)BS, NH);
  rope_kernel<<<(int)((BS * NKVH * 32 + 255) / 256), 256, 0, stream>>>(Kb, rope, (int)BS, NKVH);

  // Fused causal GQA flash attention (async K/V staging)
  attn_fused<<<dim3(BATCH * NH, SEQ / 32), 64, 0, stream>>>(Qb, Kb, Vb, AO);

  // Output projection
  gemm_xWt<<<dim3(DMODEL / 128, BS / 128), 256, 0, stream>>>(AO, Wo, out, BS, DMODEL, DMODEL);
}
